// BoundaryCycleEncoder_46815143526782
// MI455X (gfx1250) — compile-verified, matrix operand order checked by reading, not run on hardware
//
#include <hip/hip_runtime.h>
#include <hip/hip_bf16.h>

#define Bb 64
#define Ll 4096
#define Hh 128
#define NLAYERS 4
#define EPS 1e-5f

typedef __attribute__((ext_vector_type(8)))  float        v8f;
typedef __attribute__((ext_vector_type(16))) __bf16       v16bf;
typedef __attribute__((ext_vector_type(8)))  __bf16       v8bf;
typedef __attribute__((ext_vector_type(4)))  __bf16       v4bf;
typedef __attribute__((ext_vector_type(4)))  unsigned int v4u;
typedef __attribute__((ext_vector_type(8)))  int          v8i;
typedef __attribute__((ext_vector_type(4)))  int          v4i;

static __device__ __forceinline__ v16bf cat8(v8bf a, v8bf b) {
    return __builtin_shufflevector(a, b, 0,1,2,3,4,5,6,7,8,9,10,11,12,13,14,15);
}

// ---------------------------------------------------------------------------
// Weight pre-transpose + fp32->bf16:  W1[L][384][128] -> w1t[L][128(N)][384(K)]
//                                     W2[L][128][128] -> w2t[L][128(N)][128(K)]
// ---------------------------------------------------------------------------
__global__ void k_convw(const float* __restrict__ W1, const float* __restrict__ W2,
                        __bf16* __restrict__ w1t, __bf16* __restrict__ w2t) {
    int idx = blockIdx.x * 256 + threadIdx.x;
    const int N1 = NLAYERS * 384 * 128;
    const int N2 = NLAYERS * 128 * 128;
    if (idx < N1) {
        int layer = idx / (384 * 128);
        int rem   = idx - layer * 384 * 128;
        int k = rem >> 7, n = rem & 127;
        w1t[((size_t)layer * 128 + n) * 384 + k] = (__bf16)W1[idx];
    } else if (idx < N1 + N2) {
        int i2 = idx - N1;
        int layer = i2 >> 14;
        int rem   = i2 & 16383;
        int k = rem >> 7, n = rem & 127;
        w2t[((size_t)layer * 128 + n) * 128 + k] = (__bf16)W2[i2];
    }
}

// ---------------------------------------------------------------------------
// Projection: h0 = tokens @ Wp + bp  (K=8); fp32 + bf16 shadow, both dirs
// ---------------------------------------------------------------------------
__global__ void k_project(const float* __restrict__ tokens, const float* __restrict__ Wp,
                          const float* __restrict__ bp,
                          float* __restrict__ hA, float* __restrict__ hB,
                          __bf16* __restrict__ hAb, __bf16* __restrict__ hBb) {
    size_t idx = (size_t)blockIdx.x * 256 + threadIdx.x;  // < B*L*H
    int    n   = (int)(idx & 127);
    size_t row = idx >> 7;  // b*L + l
    const float* t = tokens + row * 8;
    float v = bp[n];
#pragma unroll
    for (int k = 0; k < 8; ++k) v += t[k] * Wp[k * 128 + n];
    hA[idx] = v;
    hB[idx] = v;
    __bf16 vb = (__bf16)v;
    hAb[idx] = vb;
    hBb[idx] = vb;
}

// ---------------------------------------------------------------------------
// Fused layer:  delta = relu([h<<s> blocks] @ W1 + b1) @ W2 + b2
//               h' = relu(LN(h + delta) * gamma + beta)
// Tile: 64 rows x 128 cols, 128 threads (4 waves), wave = 16 rows x 128 cols.
// dir = +1 (forward encode) / -1 (flipped-reverse encode): shift = dir*(kb-1).
// sA staged from the bf16 shadow of h via the Tensor Data Mover (TDM); TDM's
// LDS padding (4 dwords per 64 dwords) reproduces the 272-B padded row stride.
// ---------------------------------------------------------------------------
__global__ void __launch_bounds__(128)
k_layer(const float* __restrict__ hin, float* __restrict__ hout,
        const __bf16* __restrict__ hinb, __bf16* __restrict__ houtb,
        const __bf16* __restrict__ w1t, const __bf16* __restrict__ w2t,
        const float* __restrict__ b1, const float* __restrict__ b2,
        const float* __restrict__ gamma, const float* __restrict__ beta,
        int dir) {
    __shared__ __bf16 sA[66 * 136];    // rows l0-1 .. l0+64 (cyclic), bf16
    __shared__ __bf16 sAct[64 * 136];  // relu(GEMM1) activations, bf16

    const int tid    = threadIdx.x;
    const int wave   = tid >> 5;
    const int lane   = tid & 31;
    const int lrow   = lane & 15;        // row-in-tile (A) / col-in-ntile (B,C)
    const int hiHalf = lane >> 4;        // which 16-lane half
    const int asel   = hiHalf * 8;       // K-chunk select for A/B operands
    const int waveM0 = wave * 16;

    const int l0 = blockIdx.x * 64;
    const int b  = blockIdx.y;
    const float*  hrow0 = hin  + (size_t)b * Ll * Hh;
    const __bf16* hbrow = hinb + (size_t)b * Ll * Hh;

    // ---- Stage A via TDM: contiguous span of rows l0-1 .. l0+64 ------------
    {
        int startl = l0 - 1, dstrow = 0, nrows = 66;
        if (startl < 0)      { startl = 0; dstrow = 1; nrows = 65; }
        if (l0 + 64 >= Ll)   { nrows -= 1; }  // last logical row wraps

        if (tid == 0) {
            const unsigned int ldsaddr =
                (unsigned int)(unsigned long long)(const void*)&sA[0] +
                (unsigned int)dstrow * 272u;
            const unsigned long long ga =
                (unsigned long long)(const void*)(hbrow + (size_t)startl * Hh);

            v4u g0;
            g0[0] = 1u;                                   // count = 1 valid D#
            g0[1] = ldsaddr;                              // LDS byte address
            g0[2] = (unsigned int)ga;                     // global addr [31:0]
            g0[3] = (unsigned int)((ga >> 32) & 0x1FFFFFFu) | (2u << 30); // type=2

            v8i g1;
            g1[0] = (int)((2u << 16)    // data_size = 4 bytes
                        | (1u << 20)    // pad_enable
                        | (5u << 22)    // pad_interval: 64 dwords
                        | (3u << 25));  // pad_amount: 4 dwords
            g1[1] = (int)(64u << 16);   // tensor_dim0 = 64 dwords/row (lo16 in [31:16])
            g1[2] = (int)(4096u << 16); // tensor_dim0 hi=0 | tensor_dim1 lo16 = L
            g1[3] = (int)(64u << 16);   // tensor_dim1 hi=0 | tile_dim0 = 64 dwords
            g1[4] = nrows;              // tile_dim1 = rows | tile_dim2 = 0
            g1[5] = 64;                 // tensor_dim0_stride = 64 dwords (lo32)
            g1[6] = 0;                  // stride0 hi | stride1 lo
            g1[7] = 0;                  // stride1 hi
            v4i gz4 = {0, 0, 0, 0};
            v8i gz8 = {0, 0, 0, 0, 0, 0, 0, 0};

            __builtin_amdgcn_tensor_load_to_lds(g0, g1, gz4, gz4, gz8, 0);
            __builtin_amdgcn_s_wait_tensorcnt(0);
        }
        // cyclic wrap patches (first/last tile of each batch row only)
        if (l0 == 0 && tid < 32) {
            v4bf v = *(const v4bf*)&hbrow[(size_t)(Ll - 1) * Hh + tid * 4];
            *(v4bf*)&sA[0 * 136 + tid * 4] = v;
        }
        if (l0 + 64 >= Ll && tid < 32) {
            v4bf v = *(const v4bf*)&hbrow[tid * 4];
            *(v4bf*)&sA[65 * 136 + tid * 4] = v;
        }
    }
    __syncthreads();

    // ---- GEMM1: acc[M=16][N=128] = sum_kb shift(h, kb-1) @ W1_block[kb] ----
    v8f acc[8] = {};
#pragma unroll
    for (int kb = 0; kb < 3; ++kb) {
        const int s = dir * (kb - 1);
        const __bf16* arow = &sA[(waveM0 + lrow + 1 + s) * 136];
#pragma unroll
        for (int ks = 0; ks < 4; ++ks) {
            const int k0 = ks * 32 + asel;
            v16bf a = cat8(*(const v8bf*)&arow[k0], *(const v8bf*)&arow[k0 + 16]);
            const __bf16* wk = w1t + (size_t)(kb * 128 + ks * 32 + asel);
#pragma unroll
            for (int nt = 0; nt < 8; ++nt) {
                const __bf16* bptr = wk + (size_t)(nt * 16 + lrow) * 384;
                v16bf bm = cat8(*(const v8bf*)bptr, *(const v8bf*)(bptr + 16));
                acc[nt] = __builtin_amdgcn_wmma_f32_16x16x32_bf16(
                    false, a, false, bm, (short)0, acc[nt], false, false);
            }
        }
    }

    // ---- bias + ReLU -> bf16 activations in LDS ----------------------------
#pragma unroll
    for (int nt = 0; nt < 8; ++nt) {
        const int n = nt * 16 + lrow;
        const float bb = b1[n];
#pragma unroll
        for (int j = 0; j < 8; ++j) {
            float v = acc[nt][j] + bb;
            v = fmaxf(v, 0.0f);
            sAct[(waveM0 + j + 8 * hiHalf) * 136 + n] = (__bf16)v;
        }
    }
    __syncthreads();

    // ---- GEMM2: acc2 = act @ W2 --------------------------------------------
    v8f acc2[8] = {};
    const __bf16* act = &sAct[(waveM0 + lrow) * 136];
#pragma unroll
    for (int ks = 0; ks < 4; ++ks) {
        const int k0 = ks * 32 + asel;
        v16bf a = cat8(*(const v8bf*)&act[k0], *(const v8bf*)&act[k0 + 16]);
        const __bf16* wk = w2t + (size_t)(ks * 32 + asel);
#pragma unroll
        for (int nt = 0; nt < 8; ++nt) {
            const __bf16* bptr = wk + (size_t)(nt * 16 + lrow) * 128;
            v16bf bm = cat8(*(const v8bf*)bptr, *(const v8bf*)(bptr + 16));
            acc2[nt] = __builtin_amdgcn_wmma_f32_16x16x32_bf16(
                false, a, false, bm, (short)0, acc2[nt], false, false);
        }
    }

    // ---- bias2 + residual --------------------------------------------------
#pragma unroll
    for (int nt = 0; nt < 8; ++nt) {
        const int n = nt * 16 + lrow;
        const float bb = b2[n];
#pragma unroll
        for (int j = 0; j < 8; ++j) {
            const int l = l0 + waveM0 + 8 * hiHalf + j;
            acc2[nt][j] = acc2[nt][j] + bb + hrow0[(size_t)l * Hh + n];
        }
    }

    // ---- LayerNorm per row (16-lane half-wave reduction) -------------------
    float mu[8], rs[8];
#pragma unroll
    for (int j = 0; j < 8; ++j) {
        float s1 = 0.f, s2 = 0.f;
#pragma unroll
        for (int nt = 0; nt < 8; ++nt) { float v = acc2[nt][j]; s1 += v; s2 += v * v; }
#pragma unroll
        for (int m = 1; m < 16; m <<= 1) {
            s1 += __shfl_xor(s1, m);
            s2 += __shfl_xor(s2, m);
        }
        const float mean = s1 * (1.0f / 128.0f);
        const float var  = s2 * (1.0f / 128.0f) - mean * mean;
        mu[j] = mean;
        rs[j] = rsqrtf(var + EPS);
    }

    // ---- gamma/beta + ReLU + store (fp32 + bf16 shadow) --------------------
    float*  orow0 = hout  + (size_t)b * Ll * Hh;
    __bf16* obrow = houtb + (size_t)b * Ll * Hh;
#pragma unroll
    for (int nt = 0; nt < 8; ++nt) {
        const int n = nt * 16 + lrow;
        const float g = gamma[n], be = beta[n];
#pragma unroll
        for (int j = 0; j < 8; ++j) {
            const int l = l0 + waveM0 + 8 * hiHalf + j;
            float y = (acc2[nt][j] - mu[j]) * rs[j] * g + be;
            y = fmaxf(y, 0.0f);
            orow0[(size_t)l * Hh + n] = y;
            obrow[(size_t)l * Hh + n] = (__bf16)y;
        }
    }
}

// ---------------------------------------------------------------------------
// Combine h = 0.5*(hA + hB), write to d_out, and compute scores = h@Ws + bs.
// One warp per (b,l) row: lane owns 4 columns.
// ---------------------------------------------------------------------------
__global__ void k_combine_score(const float* __restrict__ hA, const float* __restrict__ hB,
                                const float* __restrict__ Ws, const float* __restrict__ bs,
                                float* __restrict__ hout, float* __restrict__ scores) {
    const int tid  = threadIdx.x;
    const int warp = tid >> 5;
    const int lane = tid & 31;
    const size_t row = (size_t)blockIdx.x * 8 + warp;  // b*L + l
    const size_t off = row * Hh + lane * 4;

    const float4 a  = *(const float4*)&hA[off];
    const float4 c  = *(const float4*)&hB[off];
    float4 hc;
    hc.x = 0.5f * (a.x + c.x);
    hc.y = 0.5f * (a.y + c.y);
    hc.z = 0.5f * (a.z + c.z);
    hc.w = 0.5f * (a.w + c.w);
    *(float4*)&hout[off] = hc;

    const float4 w = *(const float4*)&Ws[lane * 4];
    float d = hc.x * w.x + hc.y * w.y + hc.z * w.z + hc.w * w.w;
#pragma unroll
    for (int m = 16; m; m >>= 1) d += __shfl_xor(d, m);
    if (lane == 0) scores[row] = d + bs[0];
}

// ---------------------------------------------------------------------------
// Per-batch softmax over L and weighted pooling: pooled[b,:] = sum_l w_l h[b,l,:]
// ---------------------------------------------------------------------------
__global__ void k_pool(const float* __restrict__ h, const float* __restrict__ scores,
                       float* __restrict__ pooled) {
    const int b   = blockIdx.x;
    const int tid = threadIdx.x;
    __shared__ float red[256];
    __shared__ float s_mx, s_invZ;
    __shared__ float pool[8][128];

    const float* sc = scores + (size_t)b * Ll;

    float mx = -1e30f;
    for (int l = tid; l < Ll; l += 256) mx = fmaxf(mx, sc[l]);
    red[tid] = mx;
    __syncthreads();
    for (int s = 128; s; s >>= 1) {
        if (tid < s) red[tid] = fmaxf(red[tid], red[tid + s]);
        __syncthreads();
    }
    if (tid == 0) s_mx = red[0];
    __syncthreads();
    mx = s_mx;

    float sum = 0.f;
    for (int l = tid; l < Ll; l += 256) sum += __expf(sc[l] - mx);
    red[tid] = sum;
    __syncthreads();
    for (int s = 128; s; s >>= 1) {
        if (tid < s) red[tid] += red[tid + s];
        __syncthreads();
    }
    if (tid == 0) s_invZ = 1.0f / red[0];
    __syncthreads();
    const float invZ = s_invZ;

    const int warp = tid >> 5, lane = tid & 31;
    float4 acc = {0.f, 0.f, 0.f, 0.f};
    const float* hb = h + (size_t)b * Ll * Hh;
    for (int l = warp; l < Ll; l += 8) {
        const float w = __expf(sc[l] - mx) * invZ;
        const float4 hv = *(const float4*)&hb[(size_t)l * Hh + lane * 4];
        acc.x += w * hv.x; acc.y += w * hv.y; acc.z += w * hv.z; acc.w += w * hv.w;
    }
    *(float4*)&pool[warp][lane * 4] = acc;
    __syncthreads();
    if (tid < 128) {
        float s = 0.f;
#pragma unroll
        for (int w = 0; w < 8; ++w) s += pool[w][tid];
        pooled[(size_t)b * Hh + tid] = s;
    }
}

// ---------------------------------------------------------------------------
extern "C" void kernel_launch(void* const* d_in, const int* in_sizes, int n_in,
                              void* d_out, int out_size, void* d_ws, size_t ws_size,
                              hipStream_t stream) {
    (void)in_sizes; (void)n_in; (void)out_size; (void)ws_size;

    const float* tokens = (const float*)d_in[0];
    const float* Wp     = (const float*)d_in[1];
    const float* bp     = (const float*)d_in[2];
    const float* W1     = (const float*)d_in[3];
    const float* b1     = (const float*)d_in[4];
    const float* W2     = (const float*)d_in[5];
    const float* b2     = (const float*)d_in[6];
    const float* gamma  = (const float*)d_in[7];
    const float* beta   = (const float*)d_in[8];
    const float* Ws     = (const float*)d_in[9];
    const float* bs     = (const float*)d_in[10];

    const size_t S = (size_t)Bb * Ll * Hh;  // 33,554,432 elements

    float*  hA0    = (float*)d_ws;
    float*  hA1    = hA0 + S;
    float*  hB0    = hA1 + S;
    float*  hB1    = hB0 + S;
    float*  scores = hB1 + S;                        // B*L floats
    __bf16* hA0b   = (__bf16*)(scores + (size_t)Bb * Ll);
    __bf16* hA1b   = hA0b + S;
    __bf16* hB0b   = hA1b + S;
    __bf16* hB1b   = hB0b + S;
    __bf16* w1t    = hB1b + S;
    __bf16* w2t    = w1t + (size_t)NLAYERS * 128 * 384;

    float* hout   = (float*)d_out;
    float* pooled = hout + S;

    // 1) bf16 transposed weights
    {
        int total = NLAYERS * 384 * 128 + NLAYERS * 128 * 128;
        k_convw<<<(total + 255) / 256, 256, 0, stream>>>(W1, W2, w1t, w2t);
    }
    // 2) projection into both direction buffers (fp32 + bf16 shadow)
    k_project<<<(int)((S + 255) / 256), 256, 0, stream>>>(tokens, Wp, bp,
                                                          hA0, hB0, hA0b, hB0b);

    // 3) 4 layers x 2 directions (ping-pong buffers)
    float*  bufA[2]  = {hA0, hA1};
    float*  bufB[2]  = {hB0, hB1};
    __bf16* bufAb[2] = {hA0b, hA1b};
    __bf16* bufBb[2] = {hB0b, hB1b};
    dim3 grid(Ll / 64, Bb);
    for (int i = 0; i < NLAYERS; ++i) {
        const __bf16* w1L = w1t + (size_t)i * 128 * 384;
        const __bf16* w2L = w2t + (size_t)i * 128 * 128;
        k_layer<<<grid, 128, 0, stream>>>(bufA[i & 1], bufA[(i + 1) & 1],
                                          bufAb[i & 1], bufAb[(i + 1) & 1],
                                          w1L, w2L,
                                          b1 + i * Hh, b2 + i * Hh,
                                          gamma + i * Hh, beta + i * Hh, +1);
        k_layer<<<grid, 128, 0, stream>>>(bufB[i & 1], bufB[(i + 1) & 1],
                                          bufBb[i & 1], bufBb[(i + 1) & 1],
                                          w1L, w2L,
                                          b1 + i * Hh, b2 + i * Hh,
                                          gamma + i * Hh, beta + i * Hh, -1);
    }

    // 4) combine directions + scores
    k_combine_score<<<Bb * Ll / 8, 256, 0, stream>>>(bufA[0], bufB[0], Ws, bs, hout, scores);

    // 5) softmax + pooling
    k_pool<<<Bb, 256, 0, stream>>>(hout, scores, pooled);
}
